// DynamicGraphIpaFrameSeqDenoiser_49555332661530
// MI455X (gfx1250) — matrix operand order repaired and checked by reading, not compiled, not run on hardware
//
#include <hip/hip_runtime.h>
#include <hip/hip_bf16.h>
#include <math.h>

// ---------------------------------------------------------------------------
// DynamicGraphIpaFrameSeqDenoiser forward for MI455X (gfx1250, wave32, WMMA).
// All dense linears run through a 64x64-tile f16 WMMA GEMM (f32 accumulate);
// edge concat gathers (edge_feat | nf[i] | nf[j]) are fused into the GEMM's
// A-tile LDS staging so the 337MB "ei" tensor is never materialized.
// LDS tiles are padded to 40-half rows (80B: 16B-aligned, bank-stride 20) so
// WMMA fragments load as ds_load_b128; B is stored transposed for this.
// Workspace requirement: ~672 MB (see bump allocator below).
// ---------------------------------------------------------------------------

typedef __attribute__((ext_vector_type(16))) _Float16 v16h;
typedef __attribute__((ext_vector_type(8)))  _Float16 h8;
typedef __attribute__((ext_vector_type(8)))  float    v8f;

static constexpr long kN  = 8192;
static constexpr long kE  = 131072;
static constexpr long kEs = 65536;
static constexpr int  kCS = 256, kCZ = 128, kH = 8;
static constexpr int  kAA = 20, kFeat = 1408, kETH = 384;

// ---------------------------------------------------------------------------
// Tiled WMMA GEMM: C[M,N] = act( A[M,K] @ W[K,N] + bias ), A gathered/concat
// from up to 3 column segments [0,w0) [w0,w0+w1) [w0+w1,K), each with an
// optional row-index map (graph gather). Block: 256 threads = 8 wave32s;
// tile 64(M) x 64(N); K-step 32. Wave w owns rows (w>>1)*16.., cols (w&1)*32..
// (two v_wmma_f32_16x16x32_f16 per K-step).
// ---------------------------------------------------------------------------
template <int RELU>
__global__ __launch_bounds__(256) void gemm_wmma_kernel(
    const float* __restrict__ A0, int ld0, const int* __restrict__ m0,
    const float* __restrict__ A1, int ld1, const int* __restrict__ m1,
    const float* __restrict__ A2, int ld2, const int* __restrict__ m2,
    int w0, int w1,
    const float* __restrict__ W, const float* __restrict__ bias,
    float* __restrict__ C, long M, int Nc, int K)
{
  __shared__ _Float16 As[64][40] __attribute__((aligned(16)));
  __shared__ _Float16 Bt[64][40] __attribute__((aligned(16)));  // transposed [n][k]
  const int tid  = threadIdx.x;
  const long bm  = (long)blockIdx.y * 64;
  const int bn   = blockIdx.x * 64;
  const int wave = tid >> 5;
  const int lane = tid & 31;
  const int wm   = (wave >> 1) * 16;
  const int wn   = (wave & 1) * 32;

  // A staging: row is fixed for the whole K loop -> hoist the gather indices.
  const int  ar   = tid >> 2;
  const int  acb  = (tid & 3) * 8;
  const long arow = bm + ar;
  const bool arv  = arow < M;
  const float* p0 = nullptr; const float* p1 = nullptr; const float* p2 = nullptr;
  if (arv) {
    long r0 = m0 ? (long)m0[arow] : arow;
    p0 = A0 + r0 * (long)ld0;
    if (A1) { long r1 = m1 ? (long)m1[arow] : arow; p1 = A1 + r1 * (long)ld1; }
    if (A2) { long r2 = m2 ? (long)m2[arow] : arow; p2 = A2 + r2 * (long)ld2; }
  }
  // B staging coords
  const int bkk  = tid >> 3;
  const int bnb  = (tid & 7) * 8;
  const int bcol = bn + bnb;

  v8f acc0 = {}; v8f acc1 = {};

  for (int k0 = 0; k0 < K; k0 += 32) {
    { // ---- stage A 64x32: vector fast path when chunk is segment-uniform ----
      int kk = k0 + acb;
      _Float16* dst = &As[ar][acb];
      const float* src = nullptr;
      if (arv && kk + 7 < K) {
        if (kk + 7 < w0)                          src = p0 + kk;
        else if (kk >= w0 && kk + 7 < w0 + w1)    src = p1 + (kk - w0);
        else if (kk >= w0 + w1)                   src = p2 + (kk - w0 - w1);
      }
      if (src) {
        float4 x = *(const float4*)(src);
        float4 y = *(const float4*)(src + 4);
        h8 hv;
        hv[0] = (_Float16)x.x; hv[1] = (_Float16)x.y;
        hv[2] = (_Float16)x.z; hv[3] = (_Float16)x.w;
        hv[4] = (_Float16)y.x; hv[5] = (_Float16)y.y;
        hv[6] = (_Float16)y.z; hv[7] = (_Float16)y.w;
        *reinterpret_cast<h8*>(dst) = hv;
      } else {
        #pragma unroll
        for (int u = 0; u < 8; ++u) {
          int k2 = kk + u;
          float f = 0.f;
          if (arv && k2 < K) {
            if (k2 < w0)           f = p0[k2];
            else if (k2 < w0 + w1) f = p1[k2 - w0];
            else                   f = p2[k2 - w0 - w1];
          }
          dst[u] = (_Float16)f;
        }
      }
    }
    { // ---- stage B 32x64 transposed (coalesced global reads) ----
      int kg = k0 + bkk;
      if (kg < K && bcol + 7 < Nc) {
        const float* wr = W + (long)kg * Nc + bcol;
        float4 x = *(const float4*)(wr);
        float4 y = *(const float4*)(wr + 4);
        Bt[bnb + 0][bkk] = (_Float16)x.x; Bt[bnb + 1][bkk] = (_Float16)x.y;
        Bt[bnb + 2][bkk] = (_Float16)x.z; Bt[bnb + 3][bkk] = (_Float16)x.w;
        Bt[bnb + 4][bkk] = (_Float16)y.x; Bt[bnb + 5][bkk] = (_Float16)y.y;
        Bt[bnb + 6][bkk] = (_Float16)y.z; Bt[bnb + 7][bkk] = (_Float16)y.w;
      } else {
        #pragma unroll
        for (int u = 0; u < 8; ++u) {
          int col = bcol + u;
          float f = 0.f;
          if (kg < K && col < Nc) f = W[(long)kg * Nc + col];
          Bt[bnb + u][bkk] = (_Float16)f;
        }
      }
    }
    __syncthreads();

    // ---- fragments: contiguous 16B LDS loads (contiguous k order, same in A and B) ----
    int m  = wm + (lane & 15);
    int kh = (lane >> 4) * 16;          // A: lanes 0-15 K=0..15, 16-31 K=16..31
    h8 alo = *reinterpret_cast<const h8*>(&As[m][kh]);
    h8 ahi = *reinterpret_cast<const h8*>(&As[m][kh + 8]);
    v16h afrag = __builtin_shufflevector(alo, ahi,
        0,1,2,3,4,5,6,7,8,9,10,11,12,13,14,15);
    int n0 = wn + (lane & 15);
    int ks = (lane >> 4) * 16;          // B: lanes 0-15 K=0..15, 16-31 K=16..31
    h8 b0lo = *reinterpret_cast<const h8*>(&Bt[n0][ks]);
    h8 b0hi = *reinterpret_cast<const h8*>(&Bt[n0][ks + 8]);
    h8 b1lo = *reinterpret_cast<const h8*>(&Bt[n0 + 16][ks]);
    h8 b1hi = *reinterpret_cast<const h8*>(&Bt[n0 + 16][ks + 8]);
    v16h bf0 = __builtin_shufflevector(b0lo, b0hi,
        0,1,2,3,4,5,6,7,8,9,10,11,12,13,14,15);
    v16h bf1 = __builtin_shufflevector(b1lo, b1hi,
        0,1,2,3,4,5,6,7,8,9,10,11,12,13,14,15);
    acc0 = __builtin_amdgcn_wmma_f32_16x16x32_f16(false, afrag, false, bf0,
                                                  (short)0, acc0, false, false);
    acc1 = __builtin_amdgcn_wmma_f32_16x16x32_f16(false, afrag, false, bf1,
                                                  (short)0, acc1, false, false);
    __syncthreads();
  }

  // ---- epilogue: D layout = VGPR r -> row (lane>>4)*8+r, col = lane&15 ----
  {
    int nl    = lane & 15;
    int rbase = (lane >> 4) * 8;
    #pragma unroll
    for (int nt = 0; nt < 2; ++nt) {
      int col = bn + wn + nt * 16 + nl;
      if (col >= Nc) continue;
      float bb = bias ? bias[col] : 0.f;
      v8f a = nt ? acc1 : acc0;
      #pragma unroll
      for (int r = 0; r < 8; ++r) {
        long row = bm + wm + rbase + r;
        if (row < M) {
          float v = a[r] + bb;
          if (RELU) v = v > 0.f ? v : 0.f;
          C[row * (long)Nc + col] = v;
        }
      }
    }
  }
}

// ---------------------------------------------------------------------------
// LayerNorm (one wave per row, D in {128,256}); optional residual u*umask,
// optional post-multiplied row mask.
// ---------------------------------------------------------------------------
__global__ __launch_bounds__(256) void ln_kernel(
    const float* x, const float* u, const float* umask,
    const float* g, const float* b, const float* pmask,
    float* out, long M, int D)
{
  long wid = ((long)blockIdx.x * blockDim.x + threadIdx.x) >> 5;
  int lane = threadIdx.x & 31;
  if (wid >= M) return;
  int nv = D >> 5;            // <= 8
  float vals[8];
  long base = wid * (long)D;
  float um = umask ? umask[wid] : 1.f;
  float s = 0.f;
  for (int i = 0; i < nv; ++i) {
    int c = lane + (i << 5);
    float a = x[base + c];
    if (u) a += u[base + c] * um;
    vals[i] = a; s += a;
  }
  for (int off = 16; off; off >>= 1) s += __shfl_xor(s, off, 32);
  float mean = s / D;
  float vs = 0.f;
  for (int i = 0; i < nv; ++i) { float d = vals[i] - mean; vs += d * d; }
  for (int off = 16; off; off >>= 1) vs += __shfl_xor(vs, off, 32);
  float rstd = rsqrtf(vs / D + 1e-5f);
  float pm = pmask ? pmask[wid] : 1.f;
  for (int i = 0; i < nv; ++i) {
    int c = lane + (i << 5);
    out[base + c] = ((vals[i] - mean) * rstd * g[c] + b[c]) * pm;
  }
}

// ---------------------------------------------------------------------------
// Small helpers
// ---------------------------------------------------------------------------
__global__ void zero_kernel(float* p, long n) {
  long i = (long)blockIdx.x * blockDim.x + threadIdx.x;
  if (i < n) p[i] = 0.f;
}
__global__ void fillu_kernel(unsigned* p, long n, unsigned v) {
  long i = (long)blockIdx.x * blockDim.x + threadIdx.x;
  if (i < n) p[i] = v;
}

__device__ __forceinline__ unsigned f2ord(float f) {
  unsigned b = __float_as_uint(f);
  return (b & 0x80000000u) ? ~b : (b | 0x80000000u);
}
__device__ __forceinline__ float ord2f(unsigned u) {
  unsigned b = (u & 0x80000000u) ? (u ^ 0x80000000u) : ~u;
  return __uint_as_float(b);
}

// small dense linear (N in {6,8}): one thread per (row,col)
__global__ void linear_small_kernel(const float* A, const float* W, const float* bias,
                                    float* C, long M, int N, int K)
{
  long tid = (long)blockIdx.x * blockDim.x + threadIdx.x;
  if (tid >= M * N) return;
  long m = tid / N; int c = tid % N;
  float acc = bias[c];
  const float* a = A + m * (long)K;
  for (int k = 0; k < K; ++k) acc += a[k] * W[(long)k * N + c];
  C[tid] = acc;
}

// upd = (lin(nf * rnm)) * rnm   with bias inside the lin
__global__ void bb_linear_kernel(const float* nf, const float* W, const float* bias,
                                 const float* rnm, float* upd, long Nn)
{
  long tid = (long)blockIdx.x * blockDim.x + threadIdx.x;
  if (tid >= Nn * 6) return;
  long n = tid / 6; int c = tid % 6;
  float acc = 0.f;
  const float* a = nf + n * 256;
  for (int k = 0; k < 256; ++k) acc += a[k] * W[k * 6 + c];
  float r = rnm[n];
  upd[tid] = (acc * r + bias[c]) * r;
}

// local points -> global frame: out = R p + t, one thread per (n, point)
__global__ void pts_global_kernel(const float* pl, const float* R, const float* t,
                                  float* pg, long Nn, int P)
{
  long tid = (long)blockIdx.x * blockDim.x + threadIdx.x;
  if (tid >= Nn * P) return;
  long n = tid / P;
  const float* Rn = R + n * 9;
  const float* x  = pl + tid * 3;
  const float* tn = t + n * 3;
  float o0 = Rn[0]*x[0] + Rn[1]*x[1] + Rn[2]*x[2] + tn[0];
  float o1 = Rn[3]*x[0] + Rn[4]*x[1] + Rn[5]*x[2] + tn[1];
  float o2 = Rn[6]*x[0] + Rn[7]*x[1] + Rn[8]*x[2] + tn[2];
  pg[tid*3+0] = o0; pg[tid*3+1] = o1; pg[tid*3+2] = o2;
}

// per (edge, head) attention logits
__global__ void ipa_logits_kernel(const float* q, const float* k,
                                  const float* qpg, const float* kpg,
                                  const float* be, const float* hw, const float* mask,
                                  const int* ei, const int* ej, float* lg, long E)
{
  long tid = (long)blockIdx.x * blockDim.x + threadIdx.x;
  if (tid >= E * kH) return;
  long e = tid >> 3; int h = tid & 7;
  long i = ei[e], j = ej[e];
  const float* qi = q + i * 128 + h * 16;
  const float* kj = k + j * 128 + h * 16;
  float qk = 0.f;
  for (int c = 0; c < 16; ++c) qk += qi[c] * kj[c];
  qk *= 0.14433756729740643f;                    // 1/sqrt(3*16)
  const float* qp = qpg + i * 96 + h * 12;
  const float* kp = kpg + j * 96 + h * 12;
  float d2 = 0.f;
  for (int c = 0; c < 12; ++c) { float d = qp[c] - kp[c]; d2 += d * d; }
  float w = log1pf(__expf(hw[h]));               // softplus
  float pt = d2 * w * 0.13608276348795434f * -0.5f;  // sqrt(1/54)
  lg[tid] = qk + 0.5773502691896258f * be[e * 8 + h] + pt + (mask[j] - 1.f) * 1e5f;
}

__global__ void segmax_kernel(const float* lg, const int* ei, unsigned* mx, long E)
{
  long tid = (long)blockIdx.x * blockDim.x + threadIdx.x;
  if (tid >= E * kH) return;
  long e = tid >> 3; int h = tid & 7;
  atomicMax(&mx[(long)ei[e] * 8 + h], f2ord(lg[tid]));
}

__global__ void segexp_kernel(float* lg, const int* ei, const unsigned* mx,
                              float* sm, long E)
{
  long tid = (long)blockIdx.x * blockDim.x + threadIdx.x;
  if (tid >= E * kH) return;
  long e = tid >> 3; int h = tid & 7;
  long i = ei[e];
  float ex = __expf(lg[tid] - ord2f(mx[i * 8 + h]));
  lg[tid] = ex;
  atomicAdd(&sm[i * 8 + h], ex);
}

__global__ void ipa_scatter_kernel(const float* ew, const float* sm,
                                   const int* ei, const int* ej,
                                   const float* v, const float* vpg, const float* z,
                                   float* o, float* optg, float* opair, long E)
{
  long tid = (long)blockIdx.x * blockDim.x + threadIdx.x;
  if (tid >= E * kH) return;
  long e = tid >> 3; int h = tid & 7;
  long i = ei[e], j = ej[e];
  float a = ew[tid] / (sm[i * 8 + h] + 1e-9f);
  const float* vj = v + j * 128 + h * 16;
  float* oi = o + i * 128 + h * 16;
  for (int c = 0; c < 16; ++c) atomicAdd(&oi[c], a * vj[c]);
  const float* pj = vpg + j * 192 + h * 24;
  float* qi = optg + i * 192 + h * 24;
  for (int c = 0; c < 24; ++c) atomicAdd(&qi[c], a * pj[c]);
  const float* ze = z + e * 128;
  float* pi = opair + i * 1024 + h * 128;
  for (int c = 0; c < 128; ++c) atomicAdd(&pi[c], a * ze[c]);
}

// assemble feats = [o(128) | R^T(o_pt_g - t)(192) | norm(64) | o_pair(1024)]
__global__ __launch_bounds__(256) void ipa_finalize_kernel(
    const float* o, const float* optg, const float* opair,
    const float* R, const float* t, float* feats, long Nn)
{
  long n = blockIdx.x;
  int tid = threadIdx.x;
  __shared__ float sopt[192];
  __shared__ float snorm[64];
  if (tid < 64) {
    int hp = tid;
    const float* Rn = R + n * 9;
    float g0 = optg[n * 192 + hp * 3 + 0] - t[n * 3 + 0];
    float g1 = optg[n * 192 + hp * 3 + 1] - t[n * 3 + 1];
    float g2 = optg[n * 192 + hp * 3 + 2] - t[n * 3 + 2];
    float l0 = Rn[0]*g0 + Rn[3]*g1 + Rn[6]*g2;    // R^T
    float l1 = Rn[1]*g0 + Rn[4]*g1 + Rn[7]*g2;
    float l2 = Rn[2]*g0 + Rn[5]*g1 + Rn[8]*g2;
    sopt[hp*3+0] = l0; sopt[hp*3+1] = l1; sopt[hp*3+2] = l2;
    snorm[hp] = sqrtf(l0*l0 + l1*l1 + l2*l2 + 1e-8f);
  }
  __syncthreads();
  for (int c = tid; c < kFeat; c += 256) {
    float v;
    if (c < 128)      v = o[n * 128 + c];
    else if (c < 320) v = sopt[c - 128];
    else if (c < 384) v = snorm[c - 320];
    else              v = opair[n * 1024 + (c - 384)];
    feats[n * (long)kFeat + c] = v;
  }
}

// backbone frame composition
__global__ void frame_update_kernel(const float* upd, const float* R, const float* t,
                                    float* Ro, float* to, long Nn)
{
  long n = (long)blockIdx.x * blockDim.x + threadIdx.x;
  if (n >= Nn) return;
  const float* u = upd + n * 6;
  float inv = rsqrtf(1.f + u[0]*u[0] + u[1]*u[1] + u[2]*u[2]);
  float qw = inv, qx = u[0]*inv, qy = u[1]*inv, qz = u[2]*inv;
  float Q[9];
  Q[0]=1-2*(qy*qy+qz*qz); Q[1]=2*(qx*qy-qw*qz); Q[2]=2*(qx*qz+qw*qy);
  Q[3]=2*(qx*qy+qw*qz);   Q[4]=1-2*(qx*qx+qz*qz); Q[5]=2*(qy*qz-qw*qx);
  Q[6]=2*(qx*qz-qw*qy);   Q[7]=2*(qy*qz+qw*qx);   Q[8]=1-2*(qx*qx+qy*qy);
  const float* Rn = R + n * 9;
  for (int r = 0; r < 3; ++r)
    for (int c = 0; c < 3; ++c)
      Ro[n*9 + r*3 + c] = Rn[r*3+0]*Q[0*3+c] + Rn[r*3+1]*Q[1*3+c] + Rn[r*3+2]*Q[2*3+c];
  for (int r = 0; r < 3; ++r)
    to[n*3 + r] = t[n*3 + r] + Rn[r*3+0]*u[3] + Rn[r*3+1]*u[4] + Rn[r*3+2]*u[5];
}

// ET residual: g = concat(ne[si], ne[sj], sef) + ht2
__global__ void et_g_kernel(const float* ne, const int* si, const int* sj,
                            const float* sef, const float* ht2, float* g, long Es)
{
  long tid = (long)blockIdx.x * blockDim.x + threadIdx.x;
  if (tid >= Es * kETH) return;
  long e = tid / kETH; int c = tid % kETH;
  float h;
  if (c < 128)      h = ne[(long)si[e] * 128 + c];
  else if (c < 256) h = ne[(long)sj[e] * 128 + (c - 128)];
  else              h = sef[e * 128 + (c - 256)];
  g[tid] = h + ht2[tid];
}

// seq-probs transition-matrix update, one thread per node
__global__ void seq_tm_kernel(const float* sp, const float* L,
                              const unsigned char* snm, float* out, long Nn)
{
  long n = (long)blockIdx.x * blockDim.x + threadIdx.x;
  if (n >= Nn) return;
  float acc[kAA];
  if (!snm[n]) {
    for (int j = 0; j < kAA; ++j) acc[j] = sp[n * kAA + j];
  } else {
    for (int j = 0; j < kAA; ++j) acc[j] = 0.f;
    for (int i = 0; i < kAA; ++i) {
      float row[kAA]; float m = -3.4e38f;
      for (int j = 0; j < kAA; ++j) {
        row[j] = ((i == j) ? 1.f : 0.f) + L[n * 400 + i * kAA + j];
        if (row[j] > m) m = row[j];
      }
      float s = 0.f;
      for (int j = 0; j < kAA; ++j) { row[j] = __expf(row[j] - m); s += row[j]; }
      float pi = sp[n * kAA + i] / s;
      for (int j = 0; j < kAA; ++j) acc[j] += pi * row[j];
    }
  }
  for (int j = 0; j < kAA; ++j) out[n * kAA + j] = acc[j];
}

// ---------------------------------------------------------------------------
// Host-side helpers
// ---------------------------------------------------------------------------
static inline dim3 g1(long n) { return dim3((unsigned)((n + 255) / 256)); }

static void launch_gemm(hipStream_t s, bool relu,
    const float* A0, int ld0, const int* m0, int w0,
    const float* A1, int ld1, const int* m1, int w1,
    const float* A2, int ld2, const int* m2,
    const float* W, const float* bias, float* C, long M, int N, int K)
{
  dim3 grid((unsigned)((N + 63) / 64), (unsigned)((M + 63) / 64));
  if (relu)
    gemm_wmma_kernel<1><<<grid, 256, 0, s>>>(A0, ld0, m0, A1, ld1, m1, A2, ld2, m2,
                                             w0, w1, W, bias, C, M, N, K);
  else
    gemm_wmma_kernel<0><<<grid, 256, 0, s>>>(A0, ld0, m0, A1, ld1, m1, A2, ld2, m2,
                                             w0, w1, W, bias, C, M, N, K);
}
static void gemm_dense(hipStream_t s, bool relu, const float* A,
                       const float* W, const float* bias, float* C,
                       long M, int N, int K)
{
  launch_gemm(s, relu, A, K, nullptr, K, nullptr, 0, nullptr, 0,
              nullptr, 0, nullptr, W, bias, C, M, N, K);
}

struct IpaP {
  const float *Wq,*bq,*Wk,*bk,*Wv,*bv,*Wqp,*bqp,*Wkp,*bkp,*Wvp,*bvp,*Wb,*bb,*hw,*Wo,*bo;
};

// ---------------------------------------------------------------------------
extern "C" void kernel_launch(void* const* d_in, const int* in_sizes, int n_in,
                              void* d_out, int out_size, void* d_ws, size_t ws_size,
                              hipStream_t stream)
{
  (void)in_sizes; (void)n_in; (void)out_size; (void)ws_size;
  auto F = [&](int i) { return (const float*)d_in[i]; };

  // ---- input leaves (depth-first, source insertion order) ----
  const float* nf_in = F(0);
  const float* R     = F(1);
  const float* t     = F(2);
  const float* sp_in = F(3);
  const float* ef_in = F(4);
  const float* nsei  = F(5);
  const float* sef_in= F(6);
  // params: 7..94 (see mapping in analysis)
  const int* eidx = (const int*)d_in[95];
  const int *ei = eidx, *ej = eidx + kE;
  const int* sidx = (const int*)d_in[96];
  const int *si = sidx, *sj = sidx + kEs;
  const float* rmask = F(97);
  const float* rnm   = F(98);
  const unsigned char* snm = (const unsigned char*)d_in[99];

  // ---- output layout: (nf, R_new, t_new, seq_probs, ef, sef) ----
  float* out   = (float*)d_out;
  float* o_nf  = out;
  float* o_R   = o_nf + kN * kCS;
  float* o_t   = o_R  + kN * 9;
  float* o_sp  = o_t  + kN * 3;
  float* o_ef  = o_sp + kN * kAA;
  float* o_sef = o_ef + kE * kCZ;

  // ---- workspace bump allocator ----
  float* wsbase = (float*)d_ws;
  long off = 0;
  auto alloc = [&](long n) { float* p = wsbase + off; off += n; return p; };
  float* bufE0 = alloc(kE * kCZ);
  float* bufE1 = alloc(kE * kCZ);
  float* sef1  = alloc(kEs * kCZ);
  float* nf0   = alloc(kN * kCS);
  float* nf1   = alloc(kN * kCS);
  float* nf2   = alloc(kN * kCS);
  float* tmpA  = alloc(kN * kCS);
  float* tmpB  = alloc(kN * kCS);
  float* q_    = alloc(kN * 128);
  float* k_    = alloc(kN * 128);
  float* v_    = alloc(kN * 128);
  float* ptl   = alloc(kN * 192);
  float* qpg   = alloc(kN * 96);
  float* kpg   = alloc(kN * 96);
  float* vpg   = alloc(kN * 192);
  float* be    = alloc(kE * 8);
  float* lg    = alloc(kE * 8);
  float* mxf   = alloc(kN * 8);       // used as unsigned
  float* sm    = alloc(kN * 8);
  float* o_acc = alloc(kN * 128);
  float* optg  = alloc(kN * 192);
  float* opair = alloc(kN * 1024);
  float* feats = alloc(kN * kFeat);
  float* uu    = alloc(kN * kCS);
  float* ne    = alloc(kN * 128);
  float* ht1   = alloc(kEs * kETH);
  float* ht2   = alloc(kEs * kETH);
  float* gbuf  = alloc(kEs * kETH);
  float* tmL   = alloc(kN * 400);
  float* bbu   = alloc(kN * 6);

  auto mkipa = [&](int b) {
    IpaP p{F(b),F(b+1),F(b+2),F(b+3),F(b+4),F(b+5),F(b+6),F(b+7),F(b+8),
           F(b+9),F(b+10),F(b+11),F(b+12),F(b+13),F(b+14),F(b+15),F(b+16)};
    return p;
  };

  auto run_ipa = [&](const float* nfx, const float* z,
                     const int* gi, const int* gj, long E,
                     const IpaP& P, const float* lng, const float* lnb,
                     float* nfout) {
    gemm_dense(stream, false, nfx, P.Wq,  P.bq,  q_,  kN, 128, 256);
    gemm_dense(stream, false, nfx, P.Wk,  P.bk,  k_,  kN, 128, 256);
    gemm_dense(stream, false, nfx, P.Wv,  P.bv,  v_,  kN, 128, 256);
    gemm_dense(stream, false, nfx, P.Wqp, P.bqp, ptl, kN, 96, 256);
    pts_global_kernel<<<g1(kN * 32), 256, 0, stream>>>(ptl, R, t, qpg, kN, 32);
    gemm_dense(stream, false, nfx, P.Wkp, P.bkp, ptl, kN, 96, 256);
    pts_global_kernel<<<g1(kN * 32), 256, 0, stream>>>(ptl, R, t, kpg, kN, 32);
    gemm_dense(stream, false, nfx, P.Wvp, P.bvp, ptl, kN, 192, 256);
    pts_global_kernel<<<g1(kN * 64), 256, 0, stream>>>(ptl, R, t, vpg, kN, 64);
    linear_small_kernel<<<g1(E * 8), 256, 0, stream>>>(z, P.Wb, P.bb, be, E, 8, 128);
    fillu_kernel<<<g1(kN * 8), 256, 0, stream>>>((unsigned*)mxf, kN * 8, 0u);
    zero_kernel<<<g1(kN * 8), 256, 0, stream>>>(sm, kN * 8);
    zero_kernel<<<g1(kN * 128), 256, 0, stream>>>(o_acc, kN * 128);
    zero_kernel<<<g1(kN * 192), 256, 0, stream>>>(optg, kN * 192);
    zero_kernel<<<g1(kN * 1024), 256, 0, stream>>>(opair, kN * 1024);
    ipa_logits_kernel<<<g1(E * 8), 256, 0, stream>>>(q_, k_, qpg, kpg, be, P.hw,
                                                     rmask, gi, gj, lg, E);
    segmax_kernel<<<g1(E * 8), 256, 0, stream>>>(lg, gi, (unsigned*)mxf, E);
    segexp_kernel<<<g1(E * 8), 256, 0, stream>>>(lg, gi, (const unsigned*)mxf, sm, E);
    ipa_scatter_kernel<<<g1(E * 8), 256, 0, stream>>>(lg, sm, gi, gj, v_, vpg, z,
                                                      o_acc, optg, opair, E);
    ipa_finalize_kernel<<<dim3((unsigned)kN), 256, 0, stream>>>(o_acc, optg, opair,
                                                                R, t, feats, kN);
    gemm_dense(stream, false, feats, P.Wo, P.bo, uu, kN, 256, kFeat);
    ln_kernel<<<g1(kN * 32), 256, 0, stream>>>(nfx, uu, rmask, lng, lnb, nullptr,
                                               nfout, kN, 256);
  };

  // ---- 1) edge embed: fused gather-concat GEMM + 2 dense + LN ----
  launch_gemm(stream, true, ef_in, 132, nullptr, 132, nf_in, 256, ei, 256,
              nf_in, 256, ej, F(7), F(8), bufE0, kE, 128, 644);
  gemm_dense(stream, true,  bufE0, F(9),  F(10), bufE1, kE, 128, 128);
  gemm_dense(stream, false, bufE1, F(11), F(12), bufE0, kE, 128, 128);
  ln_kernel<<<g1(kE * 32), 256, 0, stream>>>(bufE0, nullptr, nullptr, F(13), F(14),
                                             nullptr, o_ef, kE, 128);

  // ---- 2) seq-edge update ----
  launch_gemm(stream, true, nsei, 132, nullptr, 132, nf_in, 256, si, 256,
              nf_in, 256, sj, F(15), F(16), bufE1, kEs, 128, 644);
  gemm_dense(stream, true,  bufE1, F(17), F(18), bufE0, kEs, 128, 128);
  gemm_dense(stream, false, bufE0, F(19), F(20), bufE1, kEs, 128, 128);
  ln_kernel<<<g1(kEs * 32), 256, 0, stream>>>(sef_in, bufE1, nullptr, F(21), F(22),
                                              nullptr, sef1, kEs, 128);

  // ---- 3) seq_probs -> node update ----
  launch_gemm(stream, true, sp_in, 20, nullptr, 20, nf_in, 256, nullptr, 256,
              nullptr, 0, nullptr, F(23), F(24), tmpA, kN, 256, 276);
  gemm_dense(stream, true,  tmpA, F(25), F(26), tmpB, kN, 256, 256);
  gemm_dense(stream, false, tmpB, F(27), F(28), tmpA, kN, 256, 256);
  ln_kernel<<<g1(kN * 32), 256, 0, stream>>>(nf_in, tmpA, nullptr, F(29), F(30),
                                             nullptr, nf0, kN, 256);

  // ---- 4) spatial IPA + 5) sequence IPA ----
  IpaP Ps = mkipa(35), Pq = mkipa(52);
  run_ipa(nf0, o_ef, ei, ej, kE,  Ps, F(31), F(32), nf1);
  run_ipa(nf1, sef1, si, sj, kEs, Pq, F(33), F(34), nf2);

  // ---- 6) node transition (residual MLP + LN, then mask) -> final nf ----
  gemm_dense(stream, true,  nf2,  F(69), F(70), tmpA, kN, 256, 256);
  gemm_dense(stream, true,  tmpA, F(71), F(72), tmpB, kN, 256, 256);
  gemm_dense(stream, false, tmpB, F(73), F(74), tmpA, kN, 256, 256);
  ln_kernel<<<g1(kN * 32), 256, 0, stream>>>(nf2, tmpA, nullptr, F(75), F(76),
                                             rmask, o_nf, kN, 256);

  // ---- 7) backbone update ----
  bb_linear_kernel<<<g1(kN * 6), 256, 0, stream>>>(o_nf, F(77), F(78), rnm, bbu, kN);
  frame_update_kernel<<<g1(kN), 256, 0, stream>>>(bbu, R, t, o_R, o_t, kN);

  // ---- 8) seq-edge transition ----
  gemm_dense(stream, false, o_nf, F(79), F(80), ne, kN, 128, 256);
  launch_gemm(stream, true, ne, 128, si, 128, ne, 128, sj, 128,
              sef1, 128, nullptr, F(81), F(82), ht1, kEs, kETH, kETH);
  gemm_dense(stream, true, ht1, F(83), F(84), ht2, kEs, kETH, kETH);
  et_g_kernel<<<g1(kEs * kETH), 256, 0, stream>>>(ne, si, sj, sef1, ht2, gbuf, kEs);
  gemm_dense(stream, false, gbuf, F(85), F(86), ht1, kEs, 128, kETH);
  ln_kernel<<<g1(kEs * 32), 256, 0, stream>>>(ht1, nullptr, nullptr, F(87), F(88),
                                              nullptr, o_sef, kEs, 128);

  // ---- 9) seq-probs transition-matrix update ----
  gemm_dense(stream, true,  o_nf, F(89), F(90), tmpA, kN, 256, 256);
  gemm_dense(stream, true,  tmpA, F(91), F(92), tmpB, kN, 256, 256);
  gemm_dense(stream, false, tmpB, F(93), F(94), tmL,  kN, 400, 256);
  seq_tm_kernel<<<g1(kN), 256, 0, stream>>>(sp_in, tmL, snm, o_sp, kN);
}
